// BertModel_20572893348748
// MI455X (gfx1250) — matrix operand order repaired and checked by reading, not compiled, not run on hardware
//
#include <hip/hip_runtime.h>
#include <hip/hip_bf16.h>

// ---------------------------------------------------------------------------
// BERT encoder forward for MI455X (gfx1250, wave32).
//  - All GEMMs: v_wmma_f32_16x16x32_bf16, fp32 accumulate.
//  - Weights converted fp32->bf16 ONCE per launch (68 MB, L2-resident).
//  - Activations converted once per use; Q/K/V/ctx flow as bf16.
//  - Tile staging uses GLOBAL_LOAD_ASYNC_TO_LDS_B128 (ASYNCcnt path),
//    with a uint4 copy fallback if the builtin is absent.
// ---------------------------------------------------------------------------

typedef __attribute__((ext_vector_type(16))) __bf16 v16bf;
typedef __attribute__((ext_vector_type(8)))  float  v8f;
typedef int v4i __attribute__((vector_size(16)));    // matches builtin param type

union Frag { v16bf v; unsigned int u[8]; };

__device__ __forceinline__ unsigned short f2bf(float f) {
    union { float f; unsigned int u; } x; x.f = f;
    unsigned int r = x.u + 0x7fffu + ((x.u >> 16) & 1u);   // round-to-nearest-even
    return (unsigned short)(r >> 16);
}

#define DD 768

// ---------- async global->LDS staging (CDNA5) ------------------------------
#if defined(__has_builtin)
#  if __has_builtin(__builtin_amdgcn_global_load_async_to_lds_b128)
#    define HAVE_ASYNC_LDS 1
#  else
#    define HAVE_ASYNC_LDS 0
#  endif
#else
#  define HAVE_ASYNC_LDS 0
#endif

#if HAVE_ASYNC_LDS
__device__ __forceinline__ void async_copy_b128(const unsigned short* g,
                                                unsigned short* l) {
    __builtin_amdgcn_global_load_async_to_lds_b128(
        (v4i __attribute__((address_space(1)))*)g,
        (v4i __attribute__((address_space(3)))*)l, 0, 0);
}
#endif

__device__ __forceinline__ void wait_async0() {
#if HAVE_ASYNC_LDS
#  if __has_builtin(__builtin_amdgcn_s_wait_asynccnt)
    __builtin_amdgcn_s_wait_asynccnt(0);
#  else
    asm volatile("s_wait_asynccnt 0x0" ::: "memory");
#  endif
#endif
}

__device__ __forceinline__ void copy_b128(const unsigned short* g,
                                          unsigned short* l) {
#if HAVE_ASYNC_LDS
    async_copy_b128(g, l);
#else
    *(uint4*)l = *(const uint4*)g;
#endif
}

// ---------------------------------------------------------------------------
// fp32 -> bf16 bulk conversion (vectorized, 8 elts/thread). n % 2048 == 0.
// ---------------------------------------------------------------------------
__global__ __launch_bounds__(256) void cvt_f32_bf16(
    const float* __restrict__ in, unsigned short* __restrict__ out, int n) {
    const int idx = (blockIdx.x * 256 + threadIdx.x) * 8;
    if (idx + 8 <= n) {
        const float4 a = *(const float4*)(in + idx);
        const float4 b = *(const float4*)(in + idx + 4);
        union { unsigned short s[8]; uint4 v; } p;
        p.s[0] = f2bf(a.x); p.s[1] = f2bf(a.y); p.s[2] = f2bf(a.z); p.s[3] = f2bf(a.w);
        p.s[4] = f2bf(b.x); p.s[5] = f2bf(b.y); p.s[6] = f2bf(b.z); p.s[7] = f2bf(b.w);
        *(uint4*)(out + idx) = p.v;
    }
}

// ---------------------------------------------------------------------------
// C[M=4096, N=768] = A_bf16[M,768] @ W_bf16[768,768] + bias.
// 256 threads (8 waves), tile 128x64, wave tile 32x32, K step 32.
// BF16OUT: write bf16 (for Q/K/V/ctx chains) else fp32 (for LN chains).
// ---------------------------------------------------------------------------
template <bool BF16OUT>
__global__ __launch_bounds__(256) void gemm_bias(
    const unsigned short* __restrict__ A, const unsigned short* __restrict__ W,
    const float* __restrict__ bias, void* __restrict__ Cout) {
    __shared__ unsigned short As[128][32];  // [m][k] (row-major, async-staged)
    __shared__ unsigned short Bs[64][36];   // [n][k] (transposed, padded)

    const int tid  = threadIdx.x;
    const int lane = tid & 31;
    const int wave = tid >> 5;
    const int wm   = wave >> 1;       // 0..3
    const int wn   = wave & 1;        // 0..1
    const int half = lane >> 4;
    const int l16  = lane & 15;
    const int n0   = blockIdx.x * 64;
    const int m0   = blockIdx.y * 128;

    v8f acc[2][2];
    acc[0][0] = {}; acc[0][1] = {}; acc[1][0] = {}; acc[1][1] = {};

    for (int kb = 0; kb < DD; kb += 32) {
        // A tile 128x32: 512 x b128 chunks, 2 per thread (async path).
        #pragma unroll
        for (int i = 0; i < 2; ++i) {
            const int c = i * 256 + tid;          // 0..511
            const int row = c >> 2, part = c & 3;
            copy_b128(A + (size_t)(m0 + row) * DD + kb + part * 8,
                      &As[row][part * 8]);
        }
        // B tile: W[kb+kk][n0+ng..+7] -> Bs[n][kk] (transposed store).
        {
            const int kk = tid >> 3;              // 0..31
            const int ng = (tid & 7) * 8;         // 0..56
            const uint4 pv = *(const uint4*)(W + (size_t)(kb + kk) * DD + n0 + ng);
            const unsigned short* pe = (const unsigned short*)&pv;
            #pragma unroll
            for (int i = 0; i < 8; ++i) Bs[ng + i][kk] = pe[i];
        }
        wait_async0();
        __syncthreads();

        Frag a[2], b[2];
        #pragma unroll
        for (int t = 0; t < 2; ++t) {
            const int mrow = wm * 32 + t * 16 + l16;
            #pragma unroll
            for (int j = 0; j < 8; ++j) {
                const int e = 2 * j;
                const int k = (half ? 8 : 0) + e + (e >= 8 ? 8 : 0);
                a[t].u[j] = *(const unsigned int*)&As[mrow][k];
            }
            const int ncol = wn * 32 + t * 16 + l16;
            #pragma unroll
            for (int j = 0; j < 8; ++j) {
                const int k = (half ? 8 : 0) + 2 * j + (j >= 4 ? 8 : 0);
                b[t].u[j] = *(const unsigned int*)&Bs[ncol][k];
            }
        }
        #pragma unroll
        for (int ii = 0; ii < 2; ++ii)
            #pragma unroll
            for (int jj = 0; jj < 2; ++jj)
                acc[ii][jj] = __builtin_amdgcn_wmma_f32_16x16x32_bf16(
                    false, a[ii].v, false, b[jj].v, (short)0, acc[ii][jj],
                    false, false);
        __syncthreads();
    }

    #pragma unroll
    for (int ii = 0; ii < 2; ++ii) {
        #pragma unroll
        for (int jj = 0; jj < 2; ++jj) {
            const int n  = n0 + wn * 32 + jj * 16 + l16;
            const float bv = bias[n];
            #pragma unroll
            for (int r = 0; r < 8; ++r) {
                const int m = m0 + wm * 32 + ii * 16 + r + half * 8;
                const float val = acc[ii][jj][r] + bv;
                if (BF16OUT)
                    ((unsigned short*)Cout)[(size_t)m * DD + n] = f2bf(val);
                else
                    ((float*)Cout)[(size_t)m * DD + n] = val;
            }
        }
    }
}

// ---------------------------------------------------------------------------
// Attention: one wave per (b, h, 16-query block). Q/K/V bf16 in, ctx bf16 out.
// ---------------------------------------------------------------------------
__global__ __launch_bounds__(32) void attention(
    const unsigned short* __restrict__ Q, const unsigned short* __restrict__ K,
    const unsigned short* __restrict__ V, const float* __restrict__ mask,
    unsigned short* __restrict__ CTX) {
    __shared__ float          sc[16][512];   // fp32 scores / probs (32 KB)
    __shared__ unsigned short qs[16][64];    // [m][k]
    __shared__ unsigned short ks[16][64];    // [key][k]
    __shared__ unsigned short vs[64][36];    // [dim][key] (transposed)

    const int lane = threadIdx.x;
    const int half = lane >> 4, l16 = lane & 15;
    const int bi = blockIdx.x;
    const int qb = bi & 31;
    const int h  = (bi >> 5) % 12;
    const int b  = bi / (32 * 12);
    const float scale = 0.125f;

    const unsigned short* qptr = Q + (size_t)(b * 512 + qb * 16) * DD + h * 64;
    const unsigned short* kptr = K + (size_t)b * 512 * DD + h * 64;
    const unsigned short* vptr = V + (size_t)b * 512 * DD + h * 64;
    const float* mptr = mask + (size_t)b * 512;

    // Stage q (16x64): 128 b128 chunks, 4 per lane.
    #pragma unroll
    for (int i = 0; i < 4; ++i) {
        const int c = i * 32 + lane;
        const int row = c >> 3, part = c & 7;
        copy_b128(qptr + (size_t)row * DD + part * 8, &qs[row][part * 8]);
    }
    wait_async0();
    __syncthreads();

    Frag aq[2];
    #pragma unroll
    for (int kk = 0; kk < 2; ++kk)
        #pragma unroll
        for (int j = 0; j < 8; ++j) {
            const int e = 2 * j;
            const int k = kk * 32 + (half ? 8 : 0) + e + (e >= 8 ? 8 : 0);
            aq[kk].u[j] = *(const unsigned int*)&qs[l16][k];
        }

    // scores = q @ k^T * scale + mask
    for (int jb = 0; jb < 32; ++jb) {
        #pragma unroll
        for (int i = 0; i < 4; ++i) {
            const int c = i * 32 + lane;
            const int row = c >> 3, part = c & 7;
            copy_b128(kptr + (size_t)(jb * 16 + row) * DD + part * 8,
                      &ks[row][part * 8]);
        }
        wait_async0();
        __syncthreads();
        Frag bk[2];
        #pragma unroll
        for (int kk = 0; kk < 2; ++kk)
            #pragma unroll
            for (int j = 0; j < 8; ++j) {
                const int k = kk * 32 + (half ? 8 : 0) + 2 * j + (j >= 4 ? 8 : 0);
                bk[kk].u[j] = *(const unsigned int*)&ks[l16][k];
            }
        v8f acc = {};
        acc = __builtin_amdgcn_wmma_f32_16x16x32_bf16(false, aq[0].v, false, bk[0].v,
                                                      (short)0, acc, false, false);
        acc = __builtin_amdgcn_wmma_f32_16x16x32_bf16(false, aq[1].v, false, bk[1].v,
                                                      (short)0, acc, false, false);
        const float mv = mptr[jb * 16 + l16];
        #pragma unroll
        for (int r = 0; r < 8; ++r)
            sc[r + half * 8][jb * 16 + l16] = acc[r] * scale + mv;
        __syncthreads();
    }

    // Row softmax: each lane handles half a row; combine via shfl_xor(16).
    {
        const int row = l16;
        const int c0  = half * 256;
        float mx = -3.0e30f;
        for (int c = 0; c < 256; ++c) mx = fmaxf(mx, sc[row][c0 + c]);
        mx = fmaxf(mx, __shfl_xor(mx, 16, 32));
        float s = 0.f;
        for (int c = 0; c < 256; ++c) s += __expf(sc[row][c0 + c] - mx);
        s += __shfl_xor(s, 16, 32);
        const float inv = 1.f / s;
        for (int c = 0; c < 256; ++c)
            sc[row][c0 + c] = __expf(sc[row][c0 + c] - mx) * inv;
    }
    __syncthreads();

    // ctx = probs @ v
    v8f co[4];
    co[0] = {}; co[1] = {}; co[2] = {}; co[3] = {};
    for (int kb = 0; kb < 16; ++kb) {
        // Stage v tile 32x64 transposed -> vs[dim][key].
        #pragma unroll
        for (int i = 0; i < 8; ++i) {
            const int c = i * 32 + lane;           // 0..255
            const int key = c >> 3, dg = (c & 7) * 8;
            const uint4 pv = *(const uint4*)(vptr + (size_t)(kb * 32 + key) * DD + dg);
            const unsigned short* pe = (const unsigned short*)&pv;
            #pragma unroll
            for (int t = 0; t < 8; ++t) vs[dg + t][key] = pe[t];
        }
        __syncthreads();
        Frag ap;
        #pragma unroll
        for (int j = 0; j < 8; ++j) {
            const int e = 2 * j;
            const int kcol = kb * 32 + (half ? 8 : 0) + e + (e >= 8 ? 8 : 0);
            const float2 p = *(const float2*)&sc[l16][kcol];
            ap.u[j] = (unsigned int)f2bf(p.x) | ((unsigned int)f2bf(p.y) << 16);
        }
        #pragma unroll
        for (int nb = 0; nb < 4; ++nb) {
            Frag bv_;
            #pragma unroll
            for (int j = 0; j < 8; ++j) {
                const int key = (half ? 8 : 0) + 2 * j + (j >= 4 ? 8 : 0);
                bv_.u[j] = *(const unsigned int*)&vs[nb * 16 + l16][key];
            }
            co[nb] = __builtin_amdgcn_wmma_f32_16x16x32_bf16(
                false, ap.v, false, bv_.v, (short)0, co[nb], false, false);
        }
        __syncthreads();
    }

    #pragma unroll
    for (int nb = 0; nb < 4; ++nb)
        #pragma unroll
        for (int r = 0; r < 8; ++r)
            CTX[(size_t)(b * 512 + qb * 16 + r + half * 8) * DD +
                h * 64 + nb * 16 + l16] = f2bf(co[nb][r]);
}

// ---------------------------------------------------------------------------
// out = LN(x + residual), row of 768 per 256-thread block.
// ---------------------------------------------------------------------------
__device__ __forceinline__ float block_reduce(float x, float* red) {
    #pragma unroll
    for (int off = 16; off > 0; off >>= 1) x += __shfl_down(x, off, 32);
    const int lane = threadIdx.x & 31, w = threadIdx.x >> 5;
    if (lane == 0) red[w] = x;
    __syncthreads();
    float t = (threadIdx.x < 8) ? red[threadIdx.x] : 0.f;
    if (w == 0) {
        #pragma unroll
        for (int off = 4; off > 0; off >>= 1) t += __shfl_down(t, off, 32);
        if (lane == 0) red[0] = t;
    }
    __syncthreads();
    const float r = red[0];
    __syncthreads();
    return r;
}

__global__ __launch_bounds__(256) void bias_res_ln(
    const float* __restrict__ X, const float* __restrict__ R,
    const float* __restrict__ w, const float* __restrict__ bb,
    float* __restrict__ O) {
    __shared__ float red[8];
    const int row = blockIdx.x;
    const int tid = threadIdx.x;
    const float* x = X + (size_t)row * DD;
    const float* r = R + (size_t)row * DD;
    float v[3];
    float s = 0.f;
    #pragma unroll
    for (int j = 0; j < 3; ++j) { v[j] = x[tid + j * 256] + r[tid + j * 256]; s += v[j]; }
    s = block_reduce(s, red);
    const float u = s * (1.f / 768.f);
    float q = 0.f;
    #pragma unroll
    for (int j = 0; j < 3; ++j) { const float d = v[j] - u; q += d * d; }
    q = block_reduce(q, red);
    const float rstd = rsqrtf(q * (1.f / 768.f) + 1e-12f);
    #pragma unroll
    for (int j = 0; j < 3; ++j) {
        const int i = tid + j * 256;
        O[(size_t)row * DD + i] = w[i] * ((v[j] - u) * rstd) + bb[i];
    }
}

// ---------------------------------------------------------------------------
extern "C" void kernel_launch(void* const* d_in, const int* in_sizes, int n_in,
                              void* d_out, int out_size, void* d_ws, size_t ws_size,
                              hipStream_t stream) {
    (void)in_sizes; (void)n_in; (void)out_size; (void)ws_size;
    const float* query   = (const float*)d_in[0];
    const float* hidden0 = (const float*)d_in[1];
    const float* mask    = (const float*)d_in[2];
    const float* Wq = (const float*)d_in[3];  const float* bq = (const float*)d_in[4];
    const float* Wk = (const float*)d_in[5];  const float* bk = (const float*)d_in[6];
    const float* Wv = (const float*)d_in[7];  const float* bv = (const float*)d_in[8];
    const float* Wo1 = (const float*)d_in[9]; const float* bo1 = (const float*)d_in[10];
    const float* l1w = (const float*)d_in[11];const float* l1b = (const float*)d_in[12];
    const float* Wo2 = (const float*)d_in[13];const float* bo2 = (const float*)d_in[14];
    const float* l2w = (const float*)d_in[15];const float* l2b = (const float*)d_in[16];

    const size_t MT  = (size_t)4096 * DD;     // activation elements
    const size_t WT  = (size_t)12 * DD * DD;  // one weight stack (12 layers)

    // fp32 scratch
    float* d1     = (float*)d_ws;
    float* d2     = d1 + MT;
    float* ao     = d2 + MT;
    float* hidbuf = ao + MT;
    // bf16 scratch
    unsigned short* qin_bf = (unsigned short*)(hidbuf + MT);
    unsigned short* hid_bf = qin_bf + MT;
    unsigned short* Qb     = hid_bf + MT;
    unsigned short* Kb     = Qb + MT;
    unsigned short* Vb     = Kb + MT;
    unsigned short* Cb     = Vb + MT;
    unsigned short* aob    = Cb + MT;
    unsigned short* Wq_bf  = aob + MT;
    unsigned short* Wk_bf  = Wq_bf + WT;
    unsigned short* Wv_bf  = Wk_bf + WT;
    unsigned short* Wo1_bf = Wv_bf + WT;
    unsigned short* Wo2_bf = Wo1_bf + WT;

    const int ACT_BLK = (int)(MT / 2048);     // 1536
    const int WGT_BLK = (int)(WT / 2048);     // 3456

    // One-time (per launch) weight + query conversion; L2-resident thereafter.
    cvt_f32_bf16<<<WGT_BLK, 256, 0, stream>>>(Wq,  Wq_bf,  (int)WT);
    cvt_f32_bf16<<<WGT_BLK, 256, 0, stream>>>(Wk,  Wk_bf,  (int)WT);
    cvt_f32_bf16<<<WGT_BLK, 256, 0, stream>>>(Wv,  Wv_bf,  (int)WT);
    cvt_f32_bf16<<<WGT_BLK, 256, 0, stream>>>(Wo1, Wo1_bf, (int)WT);
    cvt_f32_bf16<<<WGT_BLK, 256, 0, stream>>>(Wo2, Wo2_bf, (int)WT);
    cvt_f32_bf16<<<ACT_BLK, 256, 0, stream>>>(query, qin_bf, (int)MT);

    const float* hid = hidden0;
    const dim3 ggrid(12, 32), gblk(256);

    for (int l = 0; l < 12; ++l) {
        const size_t wo = (size_t)l * DD * DD;
        const size_t bo = (size_t)l * DD;
        cvt_f32_bf16<<<ACT_BLK, 256, 0, stream>>>(hid, hid_bf, (int)MT);
        gemm_bias<true><<<ggrid, gblk, 0, stream>>>(qin_bf, Wq_bf + wo, bq + bo, Qb);
        gemm_bias<true><<<ggrid, gblk, 0, stream>>>(hid_bf, Wk_bf + wo, bk + bo, Kb);
        gemm_bias<true><<<ggrid, gblk, 0, stream>>>(hid_bf, Wv_bf + wo, bv + bo, Vb);
        attention<<<8 * 12 * 32, 32, 0, stream>>>(Qb, Kb, Vb, mask, Cb);
        gemm_bias<false><<<ggrid, gblk, 0, stream>>>(Cb, Wo1_bf + wo, bo1 + bo, d1);
        bias_res_ln<<<4096, 256, 0, stream>>>(d1, hid, l1w + bo, l1b + bo, ao);
        cvt_f32_bf16<<<ACT_BLK, 256, 0, stream>>>(ao, aob, (int)MT);
        gemm_bias<false><<<ggrid, gblk, 0, stream>>>(aob, Wo2_bf + wo, bo2 + bo, d2);
        float* hout = (l == 11) ? (float*)d_out : hidbuf;
        bias_res_ln<<<4096, 256, 0, stream>>>(d2, ao, l2w + bo, l2b + bo, hout);
        hid = hout;
    }
}
